// HymbaLayer_42683384988238
// MI455X (gfx1250) — compile-verified, matrix operand order checked by reading, not run on hardware
//
#include <hip/hip_runtime.h>
#include <hip/hip_bf16.h>

// ---------------- problem constants ----------------
#define BB     2
#define TT     2048
#define CDIM   1024
#define NMETA  4
#define TEXT   (TT + NMETA)   // 2052
#define NHD    16
#define NKVH   4
#define HD     64
#define NATT   8              // attention heads used
#define NSH    8              // ssm heads
#define SS     16             // ssm state dim
#define NCH    32             // chunks (2048/64)
#define CHL    64             // chunk length

typedef __bf16 v16bf __attribute__((ext_vector_type(16)));
typedef float  v8f   __attribute__((ext_vector_type(8)));

static __device__ inline v8f wmma_bf16(v16bf a, v16bf b, v8f c) {
  // D(f32,16x16) = A(bf16,16x32) * B(bf16,32x16) + C
  return __builtin_amdgcn_wmma_f32_16x16x32_bf16(false, a, false, b, (short)0, c,
                                                 false, false);
}

// K index inside a 16x32 bf16 A-fragment (mirrored for B) for half-slot h (0..15)
// per ISA 7.12.2: VGPR v<4 -> K=2v(+hh), v>=4 -> K=16+2(v-4)(+hh); lanes>=16 add 8.
static __device__ inline int wmma_k16(int h, int lane) {
  int v = h >> 1;
  return 2 * v + ((v >= 4) ? 8 : 0) + ((lane & 16) ? 8 : 0) + (h & 1);
}

static __device__ inline v8f zero8() {
  v8f z = {0.f, 0.f, 0.f, 0.f, 0.f, 0.f, 0.f, 0.f};
  return z;
}

// CDNA5 async global->LDS DMA path (VGLOBAL op 98), tracked by ASYNCcnt.
// Per-lane: copies 16 bytes from global address vaddr into LDS at offset vdst.
static __device__ inline void async_load_b128(unsigned lds_off, const void* gaddr) {
  asm volatile("global_load_async_to_lds_b128 %0, %1, off"
               :: "v"(lds_off), "v"(gaddr)
               : "memory");
}
static __device__ inline void wait_async() {
  asm volatile("s_wait_asynccnt 0x0" ::: "memory");
}
static __device__ inline unsigned lds_addr_of(const void* p) {
  return (unsigned)(unsigned long long)(p);
}

// ---------------- prep kernels ----------------
__global__ void k_f32_to_bf16(const float* __restrict__ in, __bf16* __restrict__ out,
                              long n) {
  long i = (long)blockIdx.x * blockDim.x + threadIdx.x;
  long st = (long)gridDim.x * blockDim.x;
  for (; i < n; i += st) out[i] = (__bf16)in[i];
}

__global__ void k_build_xext(const float* __restrict__ x, const float* __restrict__ meta,
                             __bf16* __restrict__ xe) {
  long n = (long)BB * TEXT * CDIM;
  long i = (long)blockIdx.x * blockDim.x + threadIdx.x;
  long st = (long)gridDim.x * blockDim.x;
  for (; i < n; i += st) {
    long per = (long)TEXT * CDIM;
    int b = (int)(i / per);
    long r = i % per;
    int t = (int)(r / CDIM);
    int c = (int)(r % CDIM);
    float v = (t < NMETA) ? meta[t * CDIM + c]
                          : x[((long)b * TT + (t - NMETA)) * CDIM + c];
    xe[i] = (__bf16)v;
  }
}

// ---------------- generic bf16 WMMA GEMM: C(f32,MxN) = A(MxK) * W(NxK)^T --------
#define TBM 128
#define TBN 128
#define TBK 32

__global__ __launch_bounds__(256) void k_gemm_bf16(
    const __bf16* __restrict__ A, const __bf16* __restrict__ W,
    float* __restrict__ C, int M, int N, int K, long aBatch, long cBatch) {
  __shared__ __bf16 As[TBM][TBK];
  __shared__ __bf16 Ws[TBN][TBK];
  const int tid = threadIdx.x;
  const int lane = tid & 31;
  const int wave = tid >> 5;
  const int wm = wave >> 2;  // 0..1 : 64 rows
  const int wn = wave & 3;   // 0..3 : 32 cols
  const int bm = blockIdx.x * TBM;
  const int bn = blockIdx.y * TBN;
  const __bf16* Ab = A + (long)blockIdx.z * aBatch;
  float* Cb = C + (long)blockIdx.z * cBatch;
  const bool fullA = (bm + TBM) <= M;
  const bool fullW = (bn + TBN) <= N;

  v8f acc[4][2];
#pragma unroll
  for (int mt = 0; mt < 4; ++mt)
#pragma unroll
    for (int nt = 0; nt < 2; ++nt) acc[mt][nt] = zero8();

  for (int k0 = 0; k0 < K; k0 += TBK) {
    // stage A tile: 128x32 bf16 = 8192B = 512 x 16B chunks (4 per 64B row)
    if (fullA) {
#pragma unroll
      for (int c = 0; c < 2; ++c) {
        int chunk = tid + c * 256;
        int r = chunk >> 2;
        int cb = (chunk & 3) * 16;
        async_load_b128(lds_addr_of(&As[r][0]) + cb,
                        (const char*)(Ab + (long)(bm + r) * K + k0) + cb);
      }
    } else {
#pragma unroll
      for (int c = 0; c < 2; ++c) {
        int chunk = tid + c * 256;
        int r = chunk >> 2;
        int cb = (chunk & 3) * 16;
        uint4 val = make_uint4(0u, 0u, 0u, 0u);
        if (bm + r < M)
          val = *(const uint4*)((const char*)(Ab + (long)(bm + r) * K + k0) + cb);
        *(uint4*)((char*)&As[r][0] + cb) = val;
      }
    }
    // stage W tile
    if (fullW) {
#pragma unroll
      for (int c = 0; c < 2; ++c) {
        int chunk = tid + c * 256;
        int r = chunk >> 2;
        int cb = (chunk & 3) * 16;
        async_load_b128(lds_addr_of(&Ws[r][0]) + cb,
                        (const char*)(W + (long)(bn + r) * K + k0) + cb);
      }
    } else {
#pragma unroll
      for (int c = 0; c < 2; ++c) {
        int chunk = tid + c * 256;
        int r = chunk >> 2;
        int cb = (chunk & 3) * 16;
        uint4 val = make_uint4(0u, 0u, 0u, 0u);
        if (bn + r < N)
          val = *(const uint4*)((const char*)(W + (long)(bn + r) * K + k0) + cb);
        *(uint4*)((char*)&Ws[r][0] + cb) = val;
      }
    }
    // prefetch next K-slab into L2 (global_prefetch_b8)
    if (k0 + TBK < K) {
      int pr = bm + (tid & 127);
      if (pr < M) __builtin_prefetch(Ab + (long)pr * K + k0 + TBK, 0, 1);
      int wr = bn + (tid & 127);
      if (wr < N) __builtin_prefetch(W + (long)wr * K + k0 + TBK, 0, 1);
    }
    wait_async();          // our wave's async DMAs landed in LDS
    __syncthreads();       // everyone's landed

    v16bf bfrag[2];
#pragma unroll
    for (int nt = 0; nt < 2; ++nt) {
      int nb = wn * 32 + nt * 16 + (lane & 15);
#pragma unroll
      for (int h = 0; h < 16; ++h) bfrag[nt][h] = Ws[nb][wmma_k16(h, lane)];
    }
#pragma unroll
    for (int mt = 0; mt < 4; ++mt) {
      v16bf afrag;
      int mb = wm * 64 + mt * 16 + (lane & 15);
#pragma unroll
      for (int h = 0; h < 16; ++h) afrag[h] = As[mb][wmma_k16(h, lane)];
#pragma unroll
      for (int nt = 0; nt < 2; ++nt) acc[mt][nt] = wmma_bf16(afrag, bfrag[nt], acc[mt][nt]);
    }
    __syncthreads();
  }

#pragma unroll
  for (int mt = 0; mt < 4; ++mt)
#pragma unroll
    for (int nt = 0; nt < 2; ++nt)
#pragma unroll
      for (int r = 0; r < 8; ++r) {
        int row = bm + wm * 64 + mt * 16 + ((lane & 16) ? r + 8 : r);
        int col = bn + wn * 32 + nt * 16 + (lane & 15);
        if (row < M && col < N) Cb[(long)row * N + col] = acc[mt][nt][r];
      }
  __builtin_amdgcn_s_wait_tensorcnt(0);  // CDNA5 wait path (TENSORcnt==0, harmless)
}

// ---------------- RMS-norm + RoPE + layout transform ----------------
// one 32-lane wave per (b, t, sel); sel: 0..7 q heads, 8..11 k heads, 12..15 v heads
__global__ __launch_bounds__(256) void k_norm_rope(
    const float* __restrict__ qf,   // (B,TEXT,512)
    const float* __restrict__ kf,   // (B,TEXT,256)
    const float* __restrict__ vf,   // (B,TEXT,256)
    const float* __restrict__ qgain,
    __bf16* __restrict__ qa,        // (B,8,TEXT,64)
    __bf16* __restrict__ kb,        // (B,4,TEXT,64)
    __bf16* __restrict__ vb) {      // (B,4,TEXT,64)
  long gw = ((long)blockIdx.x * blockDim.x + threadIdx.x) >> 5;
  int lane = threadIdx.x & 31;
  const int per_t = NATT + NKVH + NKVH;  // 16
  long tot = (long)BB * TEXT * per_t;
  if (gw >= tot) return;
  int sel = (int)(gw % per_t);
  long bt = gw / per_t;
  int t = (int)(bt % TEXT);
  int b = (int)(bt / TEXT);

  if (sel < NATT + NKVH) {
    bool isq = sel < NATT;
    int h = isq ? sel : sel - NATT;
    const float* row = isq ? (qf + ((long)b * TEXT + t) * (NATT * HD) + h * HD)
                           : (kf + ((long)b * TEXT + t) * (NKVH * HD) + h * HD);
    float v0 = row[lane], v1 = row[lane + 32];
    float ss = v0 * v0 + v1 * v1;
#pragma unroll
    for (int d = 1; d < 32; d <<= 1) ss += __shfl_xor(ss, d, 32);
    float rr = rsqrtf(ss * (1.f / 64.f) + 1e-6f);
    // inv_freq = BASE^(-lane/32) ; ln(10000)=9.2103403...
    float f = (float)t * __expf(-(float)lane * (9.210340371976184f / 32.f));
    float sn, cs;
    __sincosf(f, &sn, &cs);
    float g = isq ? (qgain[h] * 0.125f * rr) : rr;  // fold 1/sqrt(64) into Q
    float o0 = (v0 * cs - v1 * sn) * g;
    float o1 = (v0 * sn + v1 * cs) * g;
    __bf16* dst = isq ? (qa + (((long)b * NATT + h) * TEXT + t) * HD)
                      : (kb + (((long)b * NKVH + h) * TEXT + t) * HD);
    dst[lane] = (__bf16)o0;
    dst[lane + 32] = (__bf16)o1;
  } else {
    int h = sel - NATT - NKVH;
    const float* row = vf + ((long)b * TEXT + t) * (NKVH * HD) + h * HD;
    __bf16* dst = vb + (((long)b * NKVH + h) * TEXT + t) * HD;
    dst[lane] = (__bf16)row[lane];
    dst[lane + 32] = (__bf16)row[lane + 32];
  }
}

// ---------------- flash attention (causal, GQA 2:1) ----------------
#define FBR 64
#define FBC 64
__global__ __launch_bounds__(128) void k_flash_attn(
    const __bf16* __restrict__ Q,   // (B,8,TEXT,64), pre-scaled
    const __bf16* __restrict__ Kt,  // (B,4,TEXT,64)
    const __bf16* __restrict__ Vt,  // (B,4,TEXT,64)
    const float* __restrict__ attn_scale,
    __bf16* __restrict__ Co) {      // concat (B,2048,1024), cols [0,512)
  __shared__ __bf16 Ks[FBC][HD];
  __shared__ __bf16 Vs[FBC][HD];
  __shared__ __bf16 Ps[4][16][FBC];
  const int qb = blockIdx.x, h = blockIdx.y, b = blockIdx.z;
  const int tid = threadIdx.x, lane = tid & 31, wave = tid >> 5;
  const int qbase = qb * FBR;
  const __bf16* Qh = Q + (((long)b * NATT + h) * TEXT) * HD;
  const int kvh = h >> 1;
  const __bf16* Kh = Kt + (((long)b * NKVH + kvh) * TEXT) * HD;
  const __bf16* Vh = Vt + (((long)b * NKVH + kvh) * TEXT) * HD;

  // Q A-fragments for this wave's 16 rows (K-dim = 64 -> 2 steps of 32)
  v16bf aq[2];
  int qr = qbase + wave * 16 + (lane & 15);
#pragma unroll
  for (int ks = 0; ks < 2; ++ks)
#pragma unroll
    for (int hh = 0; hh < 16; ++hh) {
      int kk = ks * 32 + wmma_k16(hh, lane);
      aq[ks][hh] = (qr < TEXT) ? Qh[(long)qr * HD + kk] : (__bf16)0.f;
    }

  v8f o[4];
#pragma unroll
  for (int nt = 0; nt < 4; ++nt) o[nt] = zero8();
  float m[8], l[8];
#pragma unroll
  for (int r = 0; r < 8; ++r) { m[r] = -1e30f; l[r] = 0.f; }

  const int jend = min(TEXT, qbase + FBR);
  for (int jb = 0; jb < jend; jb += FBC) {
    // stage K/V block: each 64x64 bf16 = 8192B = 512 x 16B chunks (8 per 128B row)
    if (jb + FBC <= TEXT) {
#pragma unroll
      for (int c = 0; c < 4; ++c) {
        int chunk = tid + c * 128;
        int r = chunk >> 3;
        int cb = (chunk & 7) * 16;
        async_load_b128(lds_addr_of(&Ks[r][0]) + cb,
                        (const char*)(Kh + (long)(jb + r) * HD) + cb);
        async_load_b128(lds_addr_of(&Vs[r][0]) + cb,
                        (const char*)(Vh + (long)(jb + r) * HD) + cb);
      }
      wait_async();
    } else {
      // tail block: zero-fill out-of-range rows (protects PV from 0*NaN)
#pragma unroll
      for (int c = 0; c < 4; ++c) {
        int chunk = tid + c * 128;
        int r = chunk >> 3;
        int cb = (chunk & 7) * 16;
        int gr = jb + r;
        uint4 kv = make_uint4(0u, 0u, 0u, 0u);
        uint4 vv = make_uint4(0u, 0u, 0u, 0u);
        if (gr < TEXT) {
          kv = *(const uint4*)((const char*)(Kh + (long)gr * HD) + cb);
          vv = *(const uint4*)((const char*)(Vh + (long)gr * HD) + cb);
        }
        *(uint4*)((char*)&Ks[r][0] + cb) = kv;
        *(uint4*)((char*)&Vs[r][0] + cb) = vv;
      }
    }
    __syncthreads();

    // S = Q K^T  (4 tiles of 16x16)
    v8f s[4];
#pragma unroll
    for (int nt = 0; nt < 4; ++nt) {
      s[nt] = zero8();
#pragma unroll
      for (int ks = 0; ks < 2; ++ks) {
        v16bf bk;
        int n = nt * 16 + (lane & 15);
#pragma unroll
        for (int hh = 0; hh < 16; ++hh) bk[hh] = Ks[n][ks * 32 + wmma_k16(hh, lane)];
        s[nt] = wmma_bf16(aq[ks], bk, s[nt]);
      }
    }
    // causal mask + online softmax (row r lives on one 16-lane half-wave)
#pragma unroll
    for (int r = 0; r < 8; ++r) {
      int row = qbase + wave * 16 + ((lane & 16) ? r + 8 : r);
      float mx = -1e30f;
#pragma unroll
      for (int nt = 0; nt < 4; ++nt) {
        int col = jb + nt * 16 + (lane & 15);
        float sv = s[nt][r];
        if (col > row || col >= TEXT) sv = -1e30f;
        s[nt][r] = sv;
        mx = fmaxf(mx, sv);
      }
#pragma unroll
      for (int d = 1; d < 16; d <<= 1) mx = fmaxf(mx, __shfl_xor(mx, d, 32));
      float nm = fmaxf(m[r], mx);
      float alpha = __expf(m[r] - nm);
      float ls = 0.f;
#pragma unroll
      for (int nt = 0; nt < 4; ++nt) {
        float p = __expf(s[nt][r] - nm);
        s[nt][r] = p;
        ls += p;
      }
#pragma unroll
      for (int d = 1; d < 16; d <<= 1) ls += __shfl_xor(ls, d, 32);
      l[r] = l[r] * alpha + ls;
      m[r] = nm;
      o[0][r] *= alpha; o[1][r] *= alpha; o[2][r] *= alpha; o[3][r] *= alpha;
    }
    // P (C-layout) -> LDS -> A-fragments
#pragma unroll
    for (int nt = 0; nt < 4; ++nt)
#pragma unroll
      for (int r = 0; r < 8; ++r) {
        int rr = (lane & 16) ? r + 8 : r;
        Ps[wave][rr][nt * 16 + (lane & 15)] = (__bf16)s[nt][r];
      }
    __syncthreads();
    // O += P V
#pragma unroll
    for (int ks = 0; ks < 2; ++ks) {
      v16bf ap;
      int rr = lane & 15;
#pragma unroll
      for (int hh = 0; hh < 16; ++hh) ap[hh] = Ps[wave][rr][ks * 32 + wmma_k16(hh, lane)];
#pragma unroll
      for (int nt = 0; nt < 4; ++nt) {
        v16bf bv;
        int n = nt * 16 + (lane & 15);
#pragma unroll
        for (int hh = 0; hh < 16; ++hh) bv[hh] = Vs[ks * 32 + wmma_k16(hh, lane)][n];
        o[nt] = wmma_bf16(ap, bv, o[nt]);
      }
    }
    __syncthreads();
  }

  const float asc = attn_scale[0];
#pragma unroll
  for (int nt = 0; nt < 4; ++nt)
#pragma unroll
    for (int r = 0; r < 8; ++r) {
      int row = qbase + wave * 16 + ((lane & 16) ? r + 8 : r);
      int tout = row - NMETA;
      if (tout >= 0 && tout < TT) {
        float ov = o[nt][r] / fmaxf(l[r], 1e-20f) * asc;
        Co[(((long)b * TT + tout) * CDIM) + h * HD + nt * 16 + (lane & 15)] = (__bf16)ov;
      }
    }
}

// ---------------- Mamba head: one block per (b, head) ----------------
__global__ __launch_bounds__(256) void k_mamba_head(
    const float* __restrict__ ssm_x,  // (B,2048,512)
    const float* __restrict__ logA,   // (8,16)
    const float* __restrict__ Bw,     // (8,16,64)
    const float* __restrict__ Cw,     // (8,16,64)
    const float* __restrict__ dtw,    // (8,16,64)
    const float* __restrict__ dtb,    // (8,16)
    const float* __restrict__ Ow,     // (8,64,16)
    const float* __restrict__ ssm_scale,
    float* __restrict__ la_ws, float* __restrict__ u_ws,
    float* __restrict__ cc_ws, float* __restrict__ h_ws,  // each (B*8,2048,16)
    __bf16* __restrict__ Co) {        // concat (B,2048,1024), cols [512,1024)
  const int bh = blockIdx.x;
  const int b = bh >> 3, h = bh & 7;
  const int tid = threadIdx.x;
  const float* xb = ssm_x + ((long)b * TT) * (NSH * HD) + h * HD;
  const long wo = (long)bh * TT * SS;
  float* la = la_ws + wo;
  float* uu = u_ws + wo;
  float* cc = cc_ws + wo;
  float* hh = h_ws + wo;

  __shared__ float sBw[SS][HD], sCw[SS][HD], sDw[SS][HD], sOw[HD][SS];
  __shared__ float sDtb[SS], sA[SS];
  for (int i = tid; i < SS * HD; i += 256) {
    sBw[i >> 6][i & 63] = Bw[h * SS * HD + i];
    sCw[i >> 6][i & 63] = Cw[h * SS * HD + i];
    sDw[i >> 6][i & 63] = dtw[h * SS * HD + i];
  }
  for (int i = tid; i < HD * SS; i += 256) sOw[i >> 4][i & 15] = Ow[h * HD * SS + i];
  if (tid < SS) {
    sDtb[tid] = dtb[h * SS + tid];
    sA[tid] = fminf(-__expf(logA[h * SS + tid]), 10.f);
  }
  __syncthreads();

  // phase 1: per-token projections
  for (int t = tid; t < TT; t += 256) {
    float xh[HD];
#pragma unroll
    for (int d = 0; d < HD; ++d) xh[d] = xb[(long)t * (NSH * HD) + d];
    for (int s = 0; s < SS; ++s) {
      float zd = 0.f, zb = 0.f, zc = 0.f;
#pragma unroll
      for (int d = 0; d < HD; ++d) {
        zd += xh[d] * sDw[s][d];
        zb += xh[d] * sBw[s][d];
        zc += xh[d] * sCw[s][d];
      }
      float dt = fminf(log1pf(__expf(zd + sDtb[s])), 1.f);  // softplus, capped
      float lav = fminf(fmaxf(dt * sA[s], -0.5f), 0.f);
      la[(long)t * SS + s] = lav;
      uu[(long)t * SS + s] = zb * dt;
      cc[(long)t * SS + s] = zc;
    }
  }
  __syncthreads();

  // phase 2: chunked scan, one lane per state (reference chunk math)
  if (tid < SS) {
    const int s = tid;
    float h0 = 0.f;
    for (int c = 0; c < NCH; ++c) {
      float lp = 0.f, acc = 0.f, hv = 0.f;
      for (int i = 0; i < CHL; ++i) {
        long t = (long)c * CHL + i;
        lp += la[t * SS + s];
        float lpc = fminf(fmaxf(lp, -30.f), 0.f);
        float p = __expf(lpc);
        acc += uu[t * SS + s] * __expf(-lpc);
        hv = p * (acc + h0);
        hh[t * SS + s] = hv;
      }
      h0 = hv;
    }
  }
  __syncthreads();

  // phase 3: y = sum(Cc*h); out = h @ Ow^T + y * x[...,0]
  const float sc = ssm_scale[0];
  for (int t = tid; t < TT; t += 256) {
    float hv[SS], cv[SS];
#pragma unroll
    for (int s = 0; s < SS; ++s) {
      hv[s] = hh[(long)t * SS + s];
      cv[s] = cc[(long)t * SS + s];
    }
    float y = 0.f;
#pragma unroll
    for (int s = 0; s < SS; ++s) y += cv[s] * hv[s];
    float yx = y * xb[(long)t * (NSH * HD) + 0];
    for (int d = 0; d < HD; ++d) {
      float ov = yx;
#pragma unroll
      for (int s = 0; s < SS; ++s) ov += hv[s] * sOw[d][s];
      Co[(((long)b * TT + t) * CDIM) + (NATT * HD) + h * HD + d] = (__bf16)(ov * sc);
    }
  }
}

// ---------------- host launcher ----------------
extern "C" void kernel_launch(void* const* d_in, const int* in_sizes, int n_in,
                              void* d_out, int out_size, void* d_ws, size_t ws_size,
                              hipStream_t stream) {
  (void)in_sizes; (void)n_in; (void)out_size; (void)ws_size;
  const float* x      = (const float*)d_in[0];
  const float* meta   = (const float*)d_in[1];
  const float* qw     = (const float*)d_in[2];
  const float* kw     = (const float*)d_in[3];
  const float* vw     = (const float*)d_in[4];
  const float* qgain  = (const float*)d_in[5];
  const float* sw     = (const float*)d_in[6];
  const float* pw     = (const float*)d_in[7];
  const float* ascale = (const float*)d_in[8];
  const float* sscale = (const float*)d_in[9];
  const float* logA   = (const float*)d_in[10];
  const float* Bw     = (const float*)d_in[11];
  const float* Cw     = (const float*)d_in[12];
  const float* dtw    = (const float*)d_in[13];
  const float* dtb    = (const float*)d_in[14];
  const float* Ow     = (const float*)d_in[15];
  float* out = (float*)d_out;

  char* ws = (char*)d_ws;
  size_t off = 0;
  auto take = [&](size_t bytes) -> void* {
    void* p = (void*)(ws + off);
    off += (bytes + 255) & ~(size_t)255;
    return p;
  };

  __bf16* XE = (__bf16*)take((size_t)BB * TEXT * CDIM * 2);      // x_ext bf16
  __bf16* QW = (__bf16*)take((size_t)512 * CDIM * 2);            // only heads 0..7 used
  __bf16* KW = (__bf16*)take((size_t)256 * CDIM * 2);
  __bf16* VW = (__bf16*)take((size_t)256 * CDIM * 2);
  __bf16* SW = (__bf16*)take((size_t)512 * CDIM * 2);
  __bf16* PW = (__bf16*)take((size_t)CDIM * CDIM * 2);
  float*  QF = (float*)take((size_t)BB * TEXT * 512 * 4);
  float*  KF = (float*)take((size_t)BB * TEXT * 256 * 4);
  float*  VF = (float*)take((size_t)BB * TEXT * 256 * 4);
  __bf16* QA = (__bf16*)take((size_t)BB * NATT * TEXT * HD * 2);
  __bf16* KB = (__bf16*)take((size_t)BB * NKVH * TEXT * HD * 2);
  __bf16* VB = (__bf16*)take((size_t)BB * NKVH * TEXT * HD * 2);
  float*  SX = (float*)take((size_t)BB * TT * 512 * 4);
  float*  LA = (float*)take((size_t)BB * NSH * TT * SS * 4);
  float*  UU = (float*)take((size_t)BB * NSH * TT * SS * 4);
  float*  CC = (float*)take((size_t)BB * NSH * TT * SS * 4);
  float*  HH = (float*)take((size_t)BB * NSH * TT * SS * 4);
  __bf16* CO = (__bf16*)take((size_t)BB * TT * CDIM * 2);        // concat bf16

  // 1. prep: x_ext + weight conversion
  k_build_xext<<<4096, 256, 0, stream>>>(x, meta, XE);
  k_f32_to_bf16<<<2048, 256, 0, stream>>>(qw, QW, (long)512 * CDIM);
  k_f32_to_bf16<<<1024, 256, 0, stream>>>(kw, KW, (long)256 * CDIM);
  k_f32_to_bf16<<<1024, 256, 0, stream>>>(vw, VW, (long)256 * CDIM);
  k_f32_to_bf16<<<2048, 256, 0, stream>>>(sw, SW, (long)512 * CDIM);
  k_f32_to_bf16<<<4096, 256, 0, stream>>>(pw, PW, (long)CDIM * CDIM);

  // 2. QKV projections (M = B*TEXT = 4104)
  const int Mext = BB * TEXT;
  dim3 gq((Mext + TBM - 1) / TBM, (512 + TBN - 1) / TBN, 1);
  k_gemm_bf16<<<gq, 256, 0, stream>>>(XE, QW, QF, Mext, 512, CDIM, 0, 0);
  dim3 gk((Mext + TBM - 1) / TBM, (256 + TBN - 1) / TBN, 1);
  k_gemm_bf16<<<gk, 256, 0, stream>>>(XE, KW, KF, Mext, 256, CDIM, 0, 0);
  k_gemm_bf16<<<gk, 256, 0, stream>>>(XE, VW, VF, Mext, 256, CDIM, 0, 0);

  // 3. norm + rope + layout (wave per (b,t,head-slot))
  {
    long waves = (long)BB * TEXT * (NATT + NKVH + NKVH);
    int blocks = (int)((waves * 32 + 255) / 256);
    k_norm_rope<<<blocks, 256, 0, stream>>>(QF, KF, VF, qgain, QA, KB, VB);
  }

  // 4. flash attention -> concat cols [0,512)
  dim3 gf((TEXT + FBR - 1) / FBR, NATT, BB);
  k_flash_attn<<<gf, 128, 0, stream>>>(QA, KB, VB, ascale, CO);

  // 5. SSM input projection per batch (A = XE skipping meta rows)
  dim3 gs((TT + TBM - 1) / TBM, (512 + TBN - 1) / TBN, BB);
  k_gemm_bf16<<<gs, 256, 0, stream>>>(XE + (size_t)NMETA * CDIM, SW, SX,
                                      TT, 512, CDIM,
                                      (long)TEXT * CDIM, (long)TT * 512);

  // 6. Mamba heads -> concat cols [512,1024)
  k_mamba_head<<<BB * NSH, 256, 0, stream>>>(SX, logA, Bw, Cw, dtw, dtb, Ow, sscale,
                                             LA, UU, CC, HH, CO);

  // 7. output projection: out = concat @ proj_w^T
  const int Mo = BB * TT;
  dim3 go((Mo + TBM - 1) / TBM, (CDIM + TBN - 1) / TBN, 1);
  k_gemm_bf16<<<go, 256, 0, stream>>>(CO, PW, out, Mo, CDIM, CDIM, 0, 0);
}
